// CausalSelfAttention_27771258536265
// MI455X (gfx1250) — matrix operand order repaired and checked by reading, not compiled, not run on hardware
//
#include <hip/hip_runtime.h>
#include <hip/hip_bf16.h>

#define TSEQ 4096
#define BATCH 2
#define NH 8
#define HD 64
#define DM 512

typedef __attribute__((ext_vector_type(16))) _Float16 v16h;
typedef __attribute__((ext_vector_type(8)))  float    v8f;
typedef int gv4i __attribute__((vector_size(16)));

union V16H { v16h v; _Float16 h[16]; };
union V8F  { v8f  v; float     f[8]; };

#define AS1 __attribute__((address_space(1)))
#define AS3 __attribute__((address_space(3)))

#if defined(__has_builtin)
#  if __has_builtin(__builtin_amdgcn_global_load_async_to_lds_b128)
#    define HAVE_ASYNC_COPY 1
#  endif
#endif
#ifndef HAVE_ASYNC_COPY
#  define HAVE_ASYNC_COPY 0
#endif

// 16-byte global -> LDS copy.  Async (ASYNCcnt-tracked) on gfx1250.
__device__ __forceinline__ void copy16_g2l(const _Float16* g, _Float16* l) {
#if HAVE_ASYNC_COPY
  __builtin_amdgcn_global_load_async_to_lds_b128(
      (AS1 gv4i*)(unsigned long long)(const void*)g,
      (AS3 gv4i*)(unsigned int)(unsigned long long)(void*)l,
      0, 0);
#else
  const uint4 v = *(const uint4*)g;
  *(uint4*)l = v;
#endif
}

__device__ __forceinline__ void async_wait0() {
#if HAVE_ASYNC_COPY
#  if __has_builtin(__builtin_amdgcn_s_wait_asynccnt)
  __builtin_amdgcn_s_wait_asynccnt(0);
#  else
  asm volatile("s_wait_asynccnt 0x0" ::: "memory");
#  endif
#endif
}

// ---------------------------------------------------------------------------
// Kernel 0: bulk f32 -> f16 conversion (read-once prepass).
// 256 threads/block, 8 elements/thread, n must be a multiple of 2048.
// ---------------------------------------------------------------------------
__global__ __launch_bounds__(256) void f32_to_f16(const float* __restrict__ src,
                                                  _Float16* __restrict__ dst) {
  const int i = (blockIdx.x * 256 + threadIdx.x) * 8;
  const float4 a = ((const float4*)(src + i))[0];
  const float4 b = ((const float4*)(src + i))[1];
  union { _Float16 h[8]; uint4 u; } o;
  o.h[0] = (_Float16)a.x; o.h[1] = (_Float16)a.y;
  o.h[2] = (_Float16)a.z; o.h[3] = (_Float16)a.w;
  o.h[4] = (_Float16)b.x; o.h[5] = (_Float16)b.y;
  o.h[6] = (_Float16)b.z; o.h[7] = (_Float16)b.w;
  *(uint4*)(dst + i) = o.u;
}

// ---------------------------------------------------------------------------
// Kernel 1: QKV projection from f16 inputs.  Y[n,o] = sum_c x[n,c]*W[o,c]
// One wave -> 16(n) x 64(o) tile.  Q gets 1/sqrt(64) folded in.
// Q,K stored f16 [B,H,T,64];  V stored TRANSPOSED f16 [B,H,64,T].
// ---------------------------------------------------------------------------
__global__ __launch_bounds__(256) void qkv_proj(const _Float16* __restrict__ xh,
                                                const _Float16* __restrict__ Wh,
                                                _Float16* __restrict__ Qh,
                                                _Float16* __restrict__ Kh,
                                                _Float16* __restrict__ VTh) {
  const int lane = threadIdx.x & 31;
  const int wave = blockIdx.x * 8 + (threadIdx.x >> 5);   // 0..12287
  const int ot = wave % 24;
  const int nt = wave / 24;
  const int n0 = nt * 16;
  const int o0 = ot * 64;
  const int m  = lane & 15;
  const int g  = lane >> 4;
  const int koA = g ? 16 : 0;
  const int koB = g ? 16 : 0;

  V8F acc[4];
#pragma unroll
  for (int t = 0; t < 4; ++t)
#pragma unroll
    for (int r = 0; r < 8; ++r) acc[t].f[r] = 0.0f;

  for (int kc = 0; kc < 16; ++kc) {
    const int c0 = kc * 32;
    const _Float16* ap = xh + (size_t)(n0 + m) * DM + c0 + koA;
    __builtin_prefetch(ap + 32, 0, 3);
    V16H a;
#pragma unroll
    for (int e = 0; e < 8; ++e) a.h[e] = ap[e];
#pragma unroll
    for (int e = 0; e < 8; ++e) a.h[8 + e] = ap[8 + e];
#pragma unroll
    for (int t = 0; t < 4; ++t) {
      const _Float16* bp = Wh + (size_t)(o0 + t * 16 + m) * DM + c0 + koB;
      V16H bm;
#pragma unroll
      for (int e = 0; e < 16; ++e) bm.h[e] = bp[e];
      acc[t].v = __builtin_amdgcn_wmma_f32_16x16x32_f16(
          false, a.v, false, bm.v, (short)0, acc[t].v, false, false);
    }
  }

  const int which = ot >> 3;
  const int head  = ot & 7;
  const float scl = (which == 0) ? 0.125f : 1.0f;   // 1/sqrt(HD) folded into Q
#pragma unroll
  for (int t = 0; t < 4; ++t)
#pragma unroll
    for (int r = 0; r < 8; ++r) {
      const int row = n0 + r + 8 * g;               // b*T + tpos
      const int b  = row >> 12;
      const int tp = row & 4095;
      const int dim = t * 16 + m;
      const _Float16 val = (_Float16)(acc[t].f[r] * scl);
      const size_t bhh = (size_t)b * NH + head;
      if (which == 0)      Qh[(bhh * TSEQ + tp) * HD + dim] = val;
      else if (which == 1) Kh[(bhh * TSEQ + tp) * HD + dim] = val;
      else                 VTh[(bhh * HD + dim) * TSEQ + tp] = val;   // V^T
    }
}

// ---------------------------------------------------------------------------
// Kernel 2: flash attention.  One block = 8 waves = 128 query rows of one
// (b,h).  K and V^T tiles (32 keys) staged into LDS, double-buffered, via
// async global->LDS copies shared by all 8 waves.
// ---------------------------------------------------------------------------
#define KT_STR 72   // halves per K-tile row   (144 B, 16B-aligned, conflict-free)
#define VT_STR 40   // halves per V^T-tile row ( 80 B, 16B-aligned, conflict-free)
#define PB_STR 40   // halves per P-buffer row

__global__ __launch_bounds__(256) void attn_fa(const _Float16* __restrict__ Qh,
                                               const _Float16* __restrict__ Kh,
                                               const _Float16* __restrict__ VTh,
                                               _Float16* __restrict__ Ah) {
  __shared__ _Float16 Kt[2][32 * KT_STR];
  __shared__ _Float16 Vt[2][64 * VT_STR];
  __shared__ _Float16 Pb[8][16 * PB_STR];

  const int tid  = threadIdx.x;
  const int lane = tid & 31;
  const int wid  = tid >> 5;
  const int bh = blockIdx.x >> 5;                  // b*NH + h
  const int qb = blockIdx.x & 31;
  const int q0blk = qb * 128;
  const int b = bh >> 3, h = bh & 7;
  const int q0 = q0blk + wid * 16;
  const int q0u = __builtin_amdgcn_readfirstlane(q0);   // wave-uniform (SGPR)

  const _Float16* Qp = Qh  + (size_t)bh * TSEQ * HD;
  const _Float16* Kp = Kh  + (size_t)bh * TSEQ * HD;
  const _Float16* Vp = VTh + (size_t)bh * HD * TSEQ;

  const int m  = lane & 15;
  const int g  = lane >> 4;
  const int koA = g ? 16 : 0;
  const int koB = g ? 16 : 0;

  // Q as two 16x32 A-fragments (head dim 64)
  v16h qa[2];
#pragma unroll
  for (int c = 0; c < 2; ++c) {
    const _Float16* p = Qp + (size_t)(q0 + m) * HD + c * 32 + koA;
    V16H a;
#pragma unroll
    for (int e = 0; e < 8; ++e) a.h[e] = p[e];
#pragma unroll
    for (int e = 0; e < 8; ++e) a.h[8 + e] = p[8 + e];
    qa[c] = a.v;
  }

  float mrow[8], lrow[8];
  V8F oacc[4];
#pragma unroll
  for (int r = 0; r < 8; ++r) { mrow[r] = -3.0e38f; lrow[r] = 0.0f; }
#pragma unroll
  for (int t = 0; t < 4; ++t)
#pragma unroll
    for (int r = 0; r < 8; ++r) oacc[t].f[r] = 0.0f;

  _Float16* pb = Pb[wid];

  // cooperative tile staging: 256 lanes x 16B cover each tile exactly
  const int kr  = tid >> 3, kc8 = (tid & 7) * 8;   // K : 32 rows x 8 chunks
  const int vr  = tid >> 2, vc8 = (tid & 3) * 8;   // V^T: 64 rows x 4 chunks

  const int nblk = (q0blk + 128) >> 5;             // number of 32-key blocks

  // preload block 0 into buffer 0
  copy16_g2l(Kp + (size_t)kr * HD + kc8,        &Kt[0][kr * KT_STR + kc8]);
  copy16_g2l(Vp + (size_t)vr * TSEQ + vc8,      &Vt[0][vr * VT_STR + vc8]);

  for (int ib = 0; ib < nblk; ++ib) {
    const int j0  = ib << 5;
    const int cur = ib & 1;
    async_wait0();
    __syncthreads();                               // tiles[cur] visible to all
    if (ib + 1 < nblk) {
      const int jn = (ib + 1) << 5;
      copy16_g2l(Kp + (size_t)(jn + kr) * HD + kc8,   &Kt[cur ^ 1][kr * KT_STR + kc8]);
      copy16_g2l(Vp + (size_t)vr * TSEQ + jn + vc8,   &Vt[cur ^ 1][vr * VT_STR + vc8]);
    }

    if (j0 <= q0u + 15) {                          // scalar causal guard
      const _Float16* kt = Kt[cur];
      const _Float16* vt = Vt[cur];

      // ---- S = Q * K^T : two 16x16 C tiles over a 32-key block ----
      V8F s[2];
#pragma unroll
      for (int t = 0; t < 2; ++t) {
#pragma unroll
        for (int r = 0; r < 8; ++r) s[t].f[r] = 0.0f;
#pragma unroll
        for (int c = 0; c < 2; ++c) {
          const _Float16* kp = kt + (t * 16 + m) * KT_STR + c * 32 + koB;
          V16H kb;
#pragma unroll
          for (int e = 0; e < 16; ++e) kb.h[e] = kp[e];
          s[t].v = __builtin_amdgcn_wmma_f32_16x16x32_f16(
              false, qa[c], false, kb.v, (short)0, s[t].v, false, false);
        }
      }
      // ---- causal mask ----
      if (j0 + 31 > q0u) {
#pragma unroll
        for (int t = 0; t < 2; ++t)
#pragma unroll
          for (int r = 0; r < 8; ++r) {
            const int q = q0 + r + 8 * g;
            const int j = j0 + t * 16 + m;
            if (j > q) s[t].f[r] = -3.0e38f;
          }
      }
      // ---- online softmax ----
      float alpha[8];
#pragma unroll
      for (int r = 0; r < 8; ++r) {
        float v = fmaxf(s[0].f[r], s[1].f[r]);
#pragma unroll
        for (int off = 1; off < 16; off <<= 1)
          v = fmaxf(v, __shfl_xor(v, off, 16));
        const float mnew = fmaxf(mrow[r], v);
        alpha[r] = __expf(mrow[r] - mnew);
        mrow[r] = mnew;
        float psum = 0.0f;
#pragma unroll
        for (int t = 0; t < 2; ++t) {
          const float p = __expf(s[t].f[r] - mnew);
          s[t].f[r] = p;
          psum += p;
        }
#pragma unroll
        for (int off = 1; off < 16; off <<= 1)
          psum += __shfl_xor(psum, off, 16);
        lrow[r] = lrow[r] * alpha[r] + psum;
      }
#pragma unroll
      for (int t = 0; t < 4; ++t)
#pragma unroll
        for (int r = 0; r < 8; ++r) oacc[t].f[r] *= alpha[r];

      // ---- P: C/D layout -> A layout via per-wave LDS staging ----
#pragma unroll
      for (int t = 0; t < 2; ++t)
#pragma unroll
        for (int r = 0; r < 8; ++r)
          pb[(r + 8 * g) * PB_STR + t * 16 + m] = (_Float16)s[t].f[r];

      V16H pa;
      {
        const _Float16* p = pb + m * PB_STR + koA;
#pragma unroll
        for (int e = 0; e < 8; ++e) pa.h[e] = p[e];
#pragma unroll
        for (int e = 0; e < 8; ++e) pa.h[8 + e] = p[8 + e];
      }
      // ---- O += P * V  (V^T rows are contiguous keys) ----
#pragma unroll
      for (int t = 0; t < 4; ++t) {
        const _Float16* vp = vt + (t * 16 + m) * VT_STR + koB;
        V16H vb;
#pragma unroll
        for (int e = 0; e < 16; ++e) vb.h[e] = vp[e];
        oacc[t].v = __builtin_amdgcn_wmma_f32_16x16x32_f16(
            false, pa.v, false, vb.v, (short)0, oacc[t].v, false, false);
      }
    }
    __syncthreads();                               // all done reading cur
  }

  // ---- epilogue: normalize by l, store f16 [B, T, 512] ----
#pragma unroll
  for (int t = 0; t < 4; ++t)
#pragma unroll
    for (int r = 0; r < 8; ++r) {
      const int q = q0 + r + 8 * g;
      const float val = oacc[t].f[r] / lrow[r];
      Ah[((size_t)b * TSEQ + q) * DM + h * HD + t * 16 + m] = (_Float16)val;
    }
}

// ---------------------------------------------------------------------------
// Kernel 3: output projection.  out[n,o] = sum_c A[n,c] * W_O[o,c]  (fp32 out)
// ---------------------------------------------------------------------------
__global__ __launch_bounds__(256) void out_proj(const _Float16* __restrict__ Ah,
                                                const _Float16* __restrict__ Woh,
                                                float* __restrict__ out) {
  const int lane = threadIdx.x & 31;
  const int wave = blockIdx.x * 8 + (threadIdx.x >> 5);   // 0..4095
  const int ot = wave & 7;
  const int nt = wave >> 3;
  const int n0 = nt * 16;
  const int o0 = ot * 64;
  const int m  = lane & 15;
  const int g  = lane >> 4;
  const int koA = g ? 16 : 0;
  const int koB = g ? 16 : 0;

  V8F acc[4];
#pragma unroll
  for (int t = 0; t < 4; ++t)
#pragma unroll
    for (int r = 0; r < 8; ++r) acc[t].f[r] = 0.0f;

  for (int kc = 0; kc < 16; ++kc) {
    const int c0 = kc * 32;
    const _Float16* ap = Ah + (size_t)(n0 + m) * DM + c0 + koA;
    __builtin_prefetch(ap + 32, 0, 3);
    V16H a;
#pragma unroll
    for (int e = 0; e < 8; ++e) a.h[e] = ap[e];
#pragma unroll
    for (int e = 0; e < 8; ++e) a.h[8 + e] = ap[8 + e];
#pragma unroll
    for (int t = 0; t < 4; ++t) {
      const _Float16* bp = Woh + (size_t)(o0 + t * 16 + m) * DM + c0 + koB;
      V16H bm;
#pragma unroll
      for (int e = 0; e < 16; ++e) bm.h[e] = bp[e];
      acc[t].v = __builtin_amdgcn_wmma_f32_16x16x32_f16(
          false, a.v, false, bm.v, (short)0, acc[t].v, false, false);
    }
  }

#pragma unroll
  for (int t = 0; t < 4; ++t)
#pragma unroll
    for (int r = 0; r < 8; ++r)
      out[(size_t)(n0 + r + 8 * g) * DM + o0 + t * 16 + m] = acc[t].f[r];
}

// ---------------------------------------------------------------------------
extern "C" void kernel_launch(void* const* d_in, const int* in_sizes, int n_in,
                              void* d_out, int out_size, void* d_ws, size_t ws_size,
                              hipStream_t stream) {
  const float* x    = (const float*)d_in[0];
  const float* Wqkv = (const float*)d_in[1];
  const float* Wo   = (const float*)d_in[2];
  float* out = (float*)d_out;

  const size_t QKV_ELEMS = (size_t)BATCH * NH * TSEQ * HD;   // 4,194,304
  const size_t X_ELEMS   = (size_t)BATCH * TSEQ * DM;        // 4,194,304
  const size_t WQKV_EL   = (size_t)3 * DM * DM;              //   786,432
  const size_t WO_EL     = (size_t)DM * DM;                  //   262,144

  _Float16* Qh    = (_Float16*)d_ws;
  _Float16* Kh    = Qh + QKV_ELEMS;
  _Float16* VTh   = Kh + QKV_ELEMS;
  _Float16* Ah    = VTh + QKV_ELEMS;
  _Float16* xh    = Ah + QKV_ELEMS;
  _Float16* Wqkvh = xh + X_ELEMS;
  _Float16* Woh   = Wqkvh + WQKV_EL;

  // prepass: convert all f32 operands to f16 once (2048 elems per block)
  f32_to_f16<<<(int)(X_ELEMS / 2048), 256, 0, stream>>>(x, xh);
  f32_to_f16<<<(int)(WQKV_EL / 2048), 256, 0, stream>>>(Wqkv, Wqkvh);
  f32_to_f16<<<(int)(WO_EL   / 2048), 256, 0, stream>>>(Wo, Woh);

  // 512 n-tiles * 24 o-tiles = 12288 waves, 8 waves/block
  qkv_proj<<<1536, 256, 0, stream>>>(xh, Wqkvh, Qh, Kh, VTh);
  // 16 (b,h) * 32 q-blocks of 128 rows
  attn_fa<<<512, 256, 0, stream>>>(Qh, Kh, VTh, Ah);
  // 512 n-tiles * 8 o-tiles = 4096 waves
  out_proj<<<512, 256, 0, stream>>>(Ah, Woh, out);
}
